// ClusterNet_61555471286940
// MI455X (gfx1250) — matrix operand order repaired and checked by reading, not compiled, not run on hardware
//
#include <hip/hip_runtime.h>
#include <hip/hip_bf16.h>
#include <math.h>

typedef __bf16 bf16_t;
typedef __attribute__((ext_vector_type(16))) __bf16 v16bf;
typedef __attribute__((ext_vector_type(8)))  float  v8f;

#define C_    512
#define S_    32
#define Q_    256
#define D_    128
#define NS_   (C_*S_)      // 16384
#define NQ_   (C_*Q_)      // 131072
#define ITERS_ 10
#define RPB_  256          // rows per column-partial block
#define PSPLIT 8
#define PCHUNK 128

// accumulator slots
#define A_VAR 0
#define A_SML 1
#define A_SMA 2
#define A_SKL 3
#define A_SKA 4
#define A_TSL 5
#define A_TSA 6

// ---------------------------------------------------------------- init
__global__ void k_zero(float* protos2, float* gq, float* gs, float* accum) {
  int i = blockIdx.x * blockDim.x + threadIdx.x;
  if (i < C_ * D_) protos2[i] = 0.f;
  if (i < C_) { gq[i] = 0.f; gs[i] = 0.f; }
  if (i < 8)  accum[i] = 0.f;
}

// ------------------------------------------------- prototypes + variance
__global__ void k_proto_var(const float* zs, float* proto, float* accum) {
  int c = blockIdx.x, d = threadIdx.x;
  const float* base = zs + (size_t)c * S_ * D_ + d;
  float s = 0.f;
#pragma unroll
  for (int i = 0; i < S_; ++i) s += base[i * D_];
  float mean = s * (1.f / S_);
  proto[c * D_ + d] = mean;
  float v = 0.f;
#pragma unroll
  for (int i = 0; i < S_; ++i) { float t = base[i * D_] - mean; v += t * t; }
  __shared__ float red[D_];
  red[d] = v; __syncthreads();
  for (int off = D_ / 2; off > 0; off >>= 1) {
    if (d < off) red[d] += red[d + off];
    __syncthreads();
  }
  if (d == 0) atomicAdd(&accum[A_VAR], red[0]);
}

// ---------------------------------------------------------- row norms
__global__ void k_rownorm(const float* X, float* out, int N) {
  int wave = (int)((blockIdx.x * blockDim.x + threadIdx.x) >> 5);
  int lane = threadIdx.x & 31;
  if (wave >= N) return;
  const float* row = X + (size_t)wave * D_;
  float s = 0.f;
#pragma unroll
  for (int t = 0; t < D_ / 32; ++t) { float v = row[t * 32 + lane]; s += v * v; }
#pragma unroll
  for (int m = 16; m >= 1; m >>= 1) s += __shfl_xor(s, m, 32);
  if (lane == 0) out[wave] = s;
}

// ---- pack Y (512x128 f32) into WMMA B-lane-layout bf16 hi/lo, x2 scale
// Layout: element ((nt*4 + kc)*32 + lane)*16 + e  holds
//   Y[nt*16 + (lane&15)][kc*32 + (lane>>4)*16 + e] * 2  (hi/lo split)
__global__ void k_packB(const float* __restrict__ Y,
                        bf16_t* __restrict__ Bh, bf16_t* __restrict__ Bl) {
  int t = blockIdx.x * blockDim.x + threadIdx.x;   // 0..4095
  int lane = t & 31;
  int kc  = (t >> 5) & 3;
  int nt  = t >> 7;
  int l15 = lane & 15;
  int hw  = lane >> 4;
  const float* rowy = Y + (size_t)(nt * 16 + l15) * D_;
  int k0 = kc * 32 + hw * 16;
  size_t base = (size_t)t * 16;
#pragma unroll
  for (int e = 0; e < 16; ++e) {
    float v = 2.0f * rowy[k0 + e];        // exact scale; folds 2*x.y
    bf16_t hi = (bf16_t)v;
    Bh[base + e] = hi;
    Bl[base + e] = (bf16_t)(v - (float)hi);
  }
}

// -------------------------------------- bf16 hi/lo conversion helper (A side)
__device__ __forceinline__ void cvt8(float4 a, float4 b, v16bf* h, v16bf* l, int e0) {
  float fv[8] = {a.x, a.y, a.z, a.w, b.x, b.y, b.z, b.w};
#pragma unroll
  for (int e = 0; e < 8; ++e) {
    bf16_t hi = (bf16_t)fv[e];
    (*h)[e0 + e] = hi;
    (*l)[e0 + e] = (bf16_t)(fv[e] - (float)hi);
  }
}

// -------- K[i,j] = 2*X_i.Y_j - |X_i|^2 - |Y_j|^2  (== -sq_dist) ------
// X: (N,128) f32; B pre-packed bf16 hi/lo (already x2 scaled).
__global__ void __launch_bounds__(256)
k_dist_wmma(const float* __restrict__ X, const float* __restrict__ xn,
            const bf16_t* __restrict__ Bh, const bf16_t* __restrict__ Bl,
            const float* __restrict__ yn, float* __restrict__ Kout, int N) {
  const int wave = threadIdx.x >> 5;
  const int lane = threadIdx.x & 31;
  const int hw   = lane >> 4;
  const int l15  = lane & 15;
  const int mBase = blockIdx.x * 128 + wave * 16;

  // A tile: 16 rows x K=128, cached in VGPRs (hi + lo), contiguous k = hw*16 + e per chunk (same as packed B)
  v16bf ah[4], al[4];
  {
    const float* rowp = X + (size_t)(mBase + l15) * D_;
#pragma unroll
    for (int kc = 0; kc < 4; ++kc) {
      int kb = kc * 32;
      int k0 = kb + hw * 16;       // elements 0..7  -> K = kb + hw*16 + e
      int k1 = kb + hw * 16 + 8;   // elements 8..15 -> K = kb + hw*16 + 8 + e
      float4 a0 = *(const float4*)(rowp + k0);
      float4 a1 = *(const float4*)(rowp + k0 + 4);
      float4 b0 = *(const float4*)(rowp + k1);
      float4 b1 = *(const float4*)(rowp + k1 + 4);
      cvt8(a0, a1, &ah[kc], &al[kc], 0);
      cvt8(b0, b1, &ah[kc], &al[kc], 8);
    }
  }
  float xnv[8];
#pragma unroll
  for (int v = 0; v < 8; ++v) xnv[v] = xn[mBase + v + hw * 8];

  for (int nt = 0; nt < C_ / 16; ++nt) {
    const int nBase = nt * 16;
    // hoist the full B tile (4 chunks x hi/lo) into one load clause
    v16bf bh[4], bl[4];
#pragma unroll
    for (int kc = 0; kc < 4; ++kc) {
      size_t boff = ((size_t)(nt * 4 + kc) * 32 + lane) * 16;
      bh[kc] = *(const v16bf*)(Bh + boff);   // 32B coalesced per lane
      bl[kc] = *(const v16bf*)(Bl + boff);
    }
    v8f c = {};
#pragma unroll
    for (int kc = 0; kc < 4; ++kc) {
      // dot ~= hi*hi + lo*hi + hi*lo (drop lo*lo, ~2^-16 rel err)
      c = __builtin_amdgcn_wmma_f32_16x16x32_bf16(false, ah[kc], false, bh[kc], (short)0, c, false, false);
      c = __builtin_amdgcn_wmma_f32_16x16x32_bf16(false, al[kc], false, bh[kc], (short)0, c, false, false);
      c = __builtin_amdgcn_wmma_f32_16x16x32_bf16(false, ah[kc], false, bl[kc], (short)0, c, false, false);
    }
    float ynv = yn[nBase + l15];
#pragma unroll
    for (int v = 0; v < 8; ++v) {
      int row = mBase + v + hw * 8;     // C/D layout: VGPR v, half-wave -> M
      Kout[(size_t)row * C_ + nBase + l15] = c[v] - xnv[v] - ynv;  // 2x folded in B
    }
  }
}

// ----------------- sinkhorn row step: f_i = log_r - LSE_j(K_ij + g_j)
__global__ void k_sk_row(const float* __restrict__ K, const float* __restrict__ g,
                         float* __restrict__ f, int N, float log_r) {
  int wave = (int)((blockIdx.x * blockDim.x + threadIdx.x) >> 5);
  int lane = threadIdx.x & 31;
  if (wave >= N) return;
  const float* row = K + (size_t)wave * C_;
  float v[16];
  float m = -INFINITY;
#pragma unroll
  for (int t = 0; t < 16; ++t) {
    int j = t * 32 + lane;
    v[t] = row[j] + g[j];
    m = fmaxf(m, v[t]);
  }
#pragma unroll
  for (int s = 16; s >= 1; s >>= 1) m = fmaxf(m, __shfl_xor(m, s, 32));
  float ssum = 0.f;
#pragma unroll
  for (int t = 0; t < 16; ++t) ssum += expf(v[t] - m);
#pragma unroll
  for (int s = 16; s >= 1; s >>= 1) ssum += __shfl_xor(ssum, s, 32);
  if (lane == 0) f[wave] = log_r - (m + logf(ssum));
}

// ------------- sinkhorn column step, stage 1: per-block partial LSE
__global__ void k_sk_col_partial(const float* __restrict__ K, const float* __restrict__ f,
                                 float* __restrict__ pmax, float* __restrict__ psum, int N) {
  int j = threadIdx.x;               // 512 columns
  int b = blockIdx.x;
  int i0 = b * RPB_;
  int i1 = i0 + RPB_; if (i1 > N) i1 = N;
  float m = -INFINITY, s = 0.f;
  for (int i = i0; i < i1; ++i) {
    float t = K[(size_t)i * C_ + j] + f[i];
    if (t > m) { s = s * expf(m - t) + 1.f; m = t; }
    else        s += expf(t - m);
  }
  pmax[b * C_ + j] = m;
  psum[b * C_ + j] = s;
}

// ------------- sinkhorn column step, stage 2: combine -> g
__global__ void k_sk_col_combine(const float* __restrict__ pmax, const float* __restrict__ psum,
                                 float* __restrict__ g, int nB, float log_c) {
  int j = threadIdx.x;
  float M = -INFINITY;
  for (int b = 0; b < nB; ++b) M = fmaxf(M, pmax[b * C_ + j]);
  float S = 0.f;
  for (int b = 0; b < nB; ++b) S += psum[b * C_ + j] * expf(pmax[b * C_ + j] - M);
  g[j] = log_c - (M + logf(S));
}

// --- fused per-row head: loss += LSE - diag ; acc += (argmax == class)
// logits = K (+ g) ; f cancels in row-softmax so it is never needed here.
__global__ void k_row_stats(const float* __restrict__ K, const float* __restrict__ g,
                            float* accum, int lossSlot, int accSlot) {
  int wave = (int)((blockIdx.x * blockDim.x + threadIdx.x) >> 5);
  int lane = threadIdx.x & 31;
  int wib  = threadIdx.x >> 5;
  const float* row = K + (size_t)wave * C_;
  float v[16];
  float m = -INFINITY; int mi = 0;
#pragma unroll
  for (int t = 0; t < 16; ++t) {
    int j = t * 32 + lane;
    float x = row[j] + (g ? g[j] : 0.f);
    v[t] = x;
    if (x > m) { m = x; mi = j; }
  }
#pragma unroll
  for (int s = 16; s >= 1; s >>= 1) {
    float om = __shfl_xor(m, s, 32);
    int   oi = __shfl_xor(mi, s, 32);
    if (om > m || (om == m && oi < mi)) { m = om; mi = oi; }
  }
  float ssum = 0.f;
#pragma unroll
  for (int t = 0; t < 16; ++t) ssum += expf(v[t] - m);
#pragma unroll
  for (int s = 16; s >= 1; s >>= 1) ssum += __shfl_xor(ssum, s, 32);
  int cls = wave / Q_;
  float diag = row[cls] + (g ? g[cls] : 0.f);
  float lse  = m + logf(ssum);
  __shared__ float sL[8], sA[8];
  if (lane == 0) { sL[wib] = lse - diag; sA[wib] = (mi == cls) ? 1.f : 0.f; }
  __syncthreads();
  if (threadIdx.x == 0) {
    float l = 0.f, a = 0.f;
#pragma unroll
    for (int w = 0; w < 8; ++w) { l += sL[w]; a += sA[w]; }
    atomicAdd(&accum[lossSlot], l);
    atomicAdd(&accum[accSlot],  a);
  }
}

// --- two-step protos: protos2[c,:] += sum_i C*exp(K_ic + f_i + g_c) * zs[i,:]
__global__ void k_protos2(const float* __restrict__ Ks, const float* __restrict__ f,
                          const float* __restrict__ g, const float* __restrict__ zs,
                          float* __restrict__ protos2) {
  int c = blockIdx.x;
  int seg = blockIdx.y;
  int d = threadIdx.x;                 // 128
  int i0 = seg * (NS_ / PSPLIT);
  int i1 = i0 + (NS_ / PSPLIT);
  float lg = logf((float)C_) + g[c];   // colsum(P)=1/C after final col update
  __shared__ float wbuf[PCHUNK];
  float acc = 0.f;
  for (int ib = i0; ib < i1; ib += PCHUNK) {
    __syncthreads();
    wbuf[d] = expf(Ks[(size_t)(ib + d) * C_ + c] + f[ib + d] + lg);
    __syncthreads();
    for (int t = 0; t < PCHUNK; ++t) acc += wbuf[t] * zs[(size_t)(ib + t) * D_ + d];
  }
  atomicAdd(&protos2[c * D_ + d], acc);
}

// ---------------------------------------------------------- finalize
__global__ void k_finalize(const float* a, float* out) {
  if (threadIdx.x == 0) {
    float invNQ = 1.f / (float)NQ_;
    out[0] = a[A_SML] * invNQ;
    out[1] = a[A_SKL] * invNQ;
    out[2] = a[A_TSL] * invNQ;
    out[3] = a[A_VAR] / (float)(C_ * S_ * D_);
    out[4] = a[A_SMA] * invNQ;
    out[5] = a[A_SKA] * invNQ;
    out[6] = a[A_TSA] * invNQ;
  }
}

extern "C" void kernel_launch(void* const* d_in, const int* in_sizes, int n_in,
                              void* d_out, int out_size, void* d_ws, size_t ws_size,
                              hipStream_t stream) {
  (void)in_sizes; (void)n_in; (void)out_size; (void)ws_size;
  const float* zs = (const float*)d_in[0];   // (512,32,128)
  const float* zq = (const float*)d_in[1];   // (512,256,128)
  float* out = (float*)d_out;

  float* w = (float*)d_ws;
  size_t off = 0;
  auto alloc = [&](size_t n) { float* p = w + off; off += n; return p; };
  float* proto   = alloc((size_t)C_ * D_);
  float* protos2 = alloc((size_t)C_ * D_);
  float* qn  = alloc(NQ_);
  float* sn  = alloc(NS_);
  float* pn  = alloc(C_);
  float* p2n = alloc(C_);
  float* f_q = alloc(NQ_);
  float* g_q = alloc(C_);
  float* f_s = alloc(NS_);
  float* g_s = alloc(C_);
  float* pmax = alloc((size_t)(NQ_ / RPB_) * C_);
  float* psum = alloc((size_t)(NQ_ / RPB_) * C_);
  float* accum = alloc(8);
  bf16_t* Bh = (bf16_t*)alloc((size_t)C_ * D_ / 2);  // 65536 bf16
  bf16_t* Bl = (bf16_t*)alloc((size_t)C_ * D_ / 2);
  float* Ks = alloc((size_t)NS_ * C_);
  float* Kq = alloc((size_t)NQ_ * C_);   // reused as Kt for the ts head

  const float log_c   = -logf((float)C_);
  const float log_r_q = -logf((float)NQ_);
  const float log_r_s = -logf((float)NS_);

  // init
  k_zero<<<(C_ * D_ + 255) / 256, 256, 0, stream>>>(protos2, g_q, g_s, accum);
  // prototypes + class variance
  k_proto_var<<<C_, D_, 0, stream>>>(zs, proto, accum);
  // norms
  k_rownorm<<<NQ_ / 8, 256, 0, stream>>>(zq, qn, NQ_);
  k_rownorm<<<NS_ / 8, 256, 0, stream>>>(zs, sn, NS_);
  k_rownorm<<<C_  / 8, 256, 0, stream>>>(proto, pn, C_);
  // pack prototypes into WMMA B layout (bf16 hi/lo, x2)
  k_packB<<<16, 256, 0, stream>>>(proto, Bh, Bl);
  // K = -sq_dist (== log_P init, REG=1)
  k_dist_wmma<<<NS_ / 128, 256, 0, stream>>>(zs, sn, Bh, Bl, pn, Ks, NS_);
  k_dist_wmma<<<NQ_ / 128, 256, 0, stream>>>(zq, qn, Bh, Bl, pn, Kq, NQ_);
  // softmax head (on raw Kq)
  k_row_stats<<<NQ_ / 8, 256, 0, stream>>>(Kq, nullptr, accum, A_SML, A_SMA);
  // sinkhorn on query dists (potential form: K + f_i + g_j, K never rewritten)
  for (int it = 0; it < ITERS_; ++it) {
    k_sk_row<<<NQ_ / 8, 256, 0, stream>>>(Kq, g_q, f_q, NQ_, log_r_q);
    k_sk_col_partial<<<NQ_ / RPB_, C_, 0, stream>>>(Kq, f_q, pmax, psum, NQ_);
    k_sk_col_combine<<<1, C_, 0, stream>>>(pmax, psum, g_q, NQ_ / RPB_, log_c);
  }
  // sinkhorn head (row-softmax of K+f+g == K+g softmax; f cancels)
  k_row_stats<<<NQ_ / 8, 256, 0, stream>>>(Kq, g_q, accum, A_SKL, A_SKA);
  // sinkhorn on support dists
  for (int it = 0; it < ITERS_; ++it) {
    k_sk_row<<<NS_ / 8, 256, 0, stream>>>(Ks, g_s, f_s, NS_, log_r_s);
    k_sk_col_partial<<<NS_ / RPB_, C_, 0, stream>>>(Ks, f_s, pmax, psum, NS_);
    k_sk_col_combine<<<1, C_, 0, stream>>>(pmax, psum, g_s, NS_ / RPB_, log_c);
  }
  // two-step prototypes: A^T @ zs_flat with A = C*exp(K+f+g)
  {
    dim3 g(C_, PSPLIT);
    k_protos2<<<g, PCHUNK, 0, stream>>>(Ks, f_s, g_s, zs, protos2);
  }
  k_rownorm<<<C_ / 8, 256, 0, stream>>>(protos2, p2n, C_);
  // ts head: pack new protos, logits into Kq buffer (dead now), fused stats
  k_packB<<<16, 256, 0, stream>>>(protos2, Bh, Bl);
  k_dist_wmma<<<NQ_ / 128, 256, 0, stream>>>(zq, qn, Bh, Bl, p2n, Kq, NQ_);
  k_row_stats<<<NQ_ / 8, 256, 0, stream>>>(Kq, nullptr, accum, A_TSL, A_TSA);
  // outputs
  k_finalize<<<1, 32, 0, stream>>>(accum, out);
}
